// LoR_Translution_16097537425419
// MI455X (gfx1250) — matrix operand order repaired and checked by reading, not compiled, not run on hardware
//
#include <hip/hip_runtime.h>
#include <stdint.h>

// ---------------------------------------------------------------------------
// Problem dims
// ---------------------------------------------------------------------------
#define Hh 14
#define Ww 14
#define HW 196
#define Nn 197            // tokens
#define NPAD 208          // tokens padded to 16
#define MPAD 224          // tokens padded to 32 (WMMA K granularity)
#define DIMc 512
#define HEADS 8
#define DH 64
#define INNER 512
#define DREL 8
#define DRH 64
#define Bb 16
#define BN (Bb * Nn)      // 3152 = 197 * 16
#define NUM_REL 732       // 27*27 + 3

typedef __attribute__((ext_vector_type(16))) __bf16 v16bf;
typedef __attribute__((ext_vector_type(8)))  __bf16 bf16x8;
typedef __attribute__((ext_vector_type(8)))  float  v8f;
typedef __attribute__((ext_vector_type(4)))  unsigned v4u;
typedef __attribute__((ext_vector_type(8)))  int  v8i;
typedef __attribute__((ext_vector_type(4)))  int  v4i;

#define DEV __device__ __forceinline__

#if __has_builtin(__builtin_amdgcn_tensor_load_to_lds)
#define HAVE_TDM 1
#else
#define HAVE_TDM 0
#endif

// ---------------------------------------------------------------------------
// CDNA5 async global->LDS copy (ASYNCcnt path, ISA 15.18.3 op 98)
// ---------------------------------------------------------------------------
DEV void async_g2l_b128(const void* gsrc, void* ldst) {
  asm volatile("global_load_async_to_lds_b128 %0, %1, off"
               :: "v"((unsigned)(uintptr_t)ldst),
                  "v"((unsigned long long)(uintptr_t)gsrc)
               : "memory");
}
DEV void wait_async0() {
#if __has_builtin(__builtin_amdgcn_s_wait_asynccnt)
  __builtin_amdgcn_s_wait_asynccnt(0);
#else
  asm volatile("s_wait_asynccnt 0x0" ::: "memory");
#endif
}
DEV void wait_tensor0() {
#if __has_builtin(__builtin_amdgcn_s_wait_tensorcnt)
  __builtin_amdgcn_s_wait_tensorcnt(0);
#else
  asm volatile("s_wait_tensorcnt 0x0" ::: "memory");
#endif
}

#if HAVE_TDM
// ---------------------------------------------------------------------------
// Tensor Data Mover: 1-D contiguous global -> LDS copy (ISA ch.8 D# layout).
// data_size = 8B units; nbytes must be a multiple of 8 (all our tiles are).
// Issue from ONE wave; completion via TENSORcnt.
// ---------------------------------------------------------------------------
DEV void tdm_copy_1d(const void* gsrc, void* ldst, unsigned nbytes) {
  unsigned n8 = nbytes >> 3;
  unsigned long long ga = (unsigned long long)(uintptr_t)gsrc;
  v4u g0;
  g0[0] = 1u;                                          // count=1 (valid), user mode
  g0[1] = (unsigned)(uintptr_t)ldst;                   // lds_addr
  g0[2] = (unsigned)(ga & 0xffffffffu);                // global_addr[31:0]
  g0[3] = (unsigned)((ga >> 32) & 0x1ffffffu) | (2u << 30);  // addr[56:32], type=2
  v8i g1;
  g1[0] = (int)(3u << 16);                             // wg_mask=0, data_size=3 (8B)
  g1[1] = (int)((n8 & 0xffffu) << 16);                 // tensor_dim0[15:0] @ bits63:48
  g1[2] = (int)((n8 >> 16) & 0xffffu) | (1 << 16);     // tensor_dim0[31:16], tensor_dim1=1
  g1[3] = (int)((n8 & 0xffffu) << 16);                 // tile_dim0 = n8 @ bits127:112
  g1[4] = 1;                                           // tile_dim1 = 1
  g1[5] = (int)n8;                                     // tensor_dim0_stride
  g1[6] = 0;
  g1[7] = 0;
  v4i z = {0, 0, 0, 0};
#if defined(__clang_major__) && (__clang_major__ >= 23)
  v8i z8 = {0, 0, 0, 0, 0, 0, 0, 0};
  __builtin_amdgcn_tensor_load_to_lds(g0, g1, z, z, z8, 0);
#else
  __builtin_amdgcn_tensor_load_to_lds(g0, g1, z, z, 0);
#endif
}
#endif // HAVE_TDM

// ---------------------------------------------------------------------------
// WMMA fragment helpers (CDNA5 wave32 layouts, ISA 7.12.2)
// A 16x32 bf16: lanes 0-15 row M=lane, K0..7 + K16..23 ; lanes 16-31 K8..15 + K24..31
// B 32x16 bf16 (loaded from Bt[col][k]): lanes 0-15 col N=lane, K0..7 + K16..23 ; lanes 16-31 K8..15 + K24..31
// C/D 16x16 f32: elem j -> row = j + 8*(lane>>4), col = lane&15
// ---------------------------------------------------------------------------
DEV v16bf load_a_frag(const __bf16* base, int lda) {
  int l = threadIdx.x & 31;
  const __bf16* p = base + (l & 15) * lda + ((l >> 4) << 3);
  bf16x8 lo = *(const bf16x8*)(p);
  bf16x8 hi = *(const bf16x8*)(p + 16);
  v16bf a;
#pragma unroll
  for (int i = 0; i < 8; ++i) { a[i] = lo[i]; a[i + 8] = hi[i]; }
  return a;
}

DEV v16bf load_b_frag(const __bf16* bt, int ldb) {
  int l = threadIdx.x & 31;
  const __bf16* p = bt + (l & 15) * ldb + ((l >> 4) << 3);
  bf16x8 lo = *(const bf16x8*)(p);
  bf16x8 hi = *(const bf16x8*)(p + 16);
  v16bf b;
#pragma unroll
  for (int i = 0; i < 8; ++i) { b[i] = lo[i]; b[i + 8] = hi[i]; }
  return b;
}

DEV v8f wmma_bf16(v16bf a, v16bf b, v8f c) {
  return __builtin_amdgcn_wmma_f32_16x16x32_bf16(false, a, false, b, (short)0, c,
                                                 false, false);
}

// relative-position index (reference _make_index_map, computed analytically)
DEV int rel_index(int n, int m) {
  if (n == 0 && m == 0) return 729;
  if (n == 0) return 730;
  if (m == 0) return 731;
  int x = (n - 1) / Ww, y = (n - 1) % Ww;
  int i = (m - 1) / Ww, j = (m - 1) % Ww;
  return (x - i + Hh - 1) * (2 * Ww - 1) + (y - j + Ww - 1);
}

// ---------------------------------------------------------------------------
// 1) batched fp32 -> bf16 transpose:  out[b][c][r] = in[b][r][c]
// ---------------------------------------------------------------------------
__global__ void k_transpose(const float* __restrict__ in, __bf16* __restrict__ out,
                            int batch, int R, int C) {
  long total = (long)batch * R * C;
  for (long i = blockIdx.x * (long)blockDim.x + threadIdx.x; i < total;
       i += (long)gridDim.x * blockDim.x) {
    long b = i / ((long)R * C);
    long rem = i - b * (long)R * C;
    int r = (int)(rem / C), c = (int)(rem % C);
    out[b * (long)R * C + (long)c * R + r] = (__bf16)in[i];
  }
}

// ---------------------------------------------------------------------------
// 2) LayerNorm over DIM=512, output bf16 row-major [BN][512]
// ---------------------------------------------------------------------------
__global__ void k_layernorm(const float* __restrict__ x, const float* __restrict__ g,
                            const float* __restrict__ be, __bf16* __restrict__ xn) {
  int row = blockIdx.x;               // 0..BN-1
  const float* xr = x + (long)row * DIMc;
  int t = threadIdx.x;                // 256 threads
  float v0 = xr[t], v1 = xr[t + 256];
  __shared__ float red[8];
  __shared__ float stat[2];
  int wv = t >> 5, ln = t & 31;
  float s = v0 + v1;
#pragma unroll
  for (int o = 16; o > 0; o >>= 1) s += __shfl_xor(s, o);
  if (ln == 0) red[wv] = s;
  __syncthreads();
  if (t == 0) {
    float tot = 0.f;
#pragma unroll
    for (int i = 0; i < 8; ++i) tot += red[i];
    stat[0] = tot * (1.f / DIMc);
  }
  __syncthreads();
  float mu = stat[0];
  float d0 = v0 - mu, d1 = v1 - mu;
  float q = d0 * d0 + d1 * d1;
#pragma unroll
  for (int o = 16; o > 0; o >>= 1) q += __shfl_xor(q, o);
  __syncthreads();
  if (ln == 0) red[wv] = q;
  __syncthreads();
  if (t == 0) {
    float tot = 0.f;
#pragma unroll
    for (int i = 0; i < 8; ++i) tot += red[i];
    stat[1] = rsqrtf(tot * (1.f / DIMc) + 1e-5f);
  }
  __syncthreads();
  float rs = stat[1];
  xn[(long)row * DIMc + t]       = (__bf16)(d0 * rs * g[t] + be[t]);
  xn[(long)row * DIMc + t + 256] = (__bf16)(d1 * rs * g[t + 256] + be[t + 256]);
}

// ---------------------------------------------------------------------------
// 3) qkv GEMM: [3152x512] @ [512x1536], software-pipelined fragments.
//    Epilogue scatters to q_pre/k_pre ([b][h][NPAD][64]) and v1t ([b][h][64][MPAD])
// ---------------------------------------------------------------------------
__global__ void k_gemm_qkv(const __bf16* __restrict__ A, const __bf16* __restrict__ Bt,
                           __bf16* __restrict__ qp, __bf16* __restrict__ kp,
                           __bf16* __restrict__ v1t) {
  int wave = (blockIdx.x * blockDim.x + threadIdx.x) >> 5;
  int lane = threadIdx.x & 31;
  const int MT = BN / 16, NT = (3 * INNER) / 64;     // 197 x 24
  if (wave >= MT * NT) return;
  int mt = wave / NT, nt = wave % NT;
  int row0 = mt * 16, col0 = nt * 64;
  v8f acc0 = {}, acc1 = {}, acc2 = {}, acc3 = {};
  const __bf16* arow = A + (long)row0 * DIMc;
  const __bf16* bp = Bt + (long)col0 * DIMc;
  v16bf a  = load_a_frag(arow, DIMc);
  v16bf b0 = load_b_frag(bp + 0L * 16 * DIMc, DIMc);
  v16bf b1 = load_b_frag(bp + 1L * 16 * DIMc, DIMc);
  v16bf b2 = load_b_frag(bp + 2L * 16 * DIMc, DIMc);
  v16bf b3 = load_b_frag(bp + 3L * 16 * DIMc, DIMc);
#pragma unroll
  for (int k = 0; k < DIMc; k += 32) {
    v16bf an = a, c0 = b0, c1 = b1, c2 = b2, c3 = b3;
    if (k + 32 < DIMc) {
      __builtin_prefetch(arow + k + 96, 0, 0);
      an = load_a_frag(arow + k + 32, DIMc);
      c0 = load_b_frag(bp + 0L * 16 * DIMc + k + 32, DIMc);
      c1 = load_b_frag(bp + 1L * 16 * DIMc + k + 32, DIMc);
      c2 = load_b_frag(bp + 2L * 16 * DIMc + k + 32, DIMc);
      c3 = load_b_frag(bp + 3L * 16 * DIMc + k + 32, DIMc);
    }
    acc0 = wmma_bf16(a, b0, acc0);
    acc1 = wmma_bf16(a, b1, acc1);
    acc2 = wmma_bf16(a, b2, acc2);
    acc3 = wmma_bf16(a, b3, acc3);
    a = an; b0 = c0; b1 = c1; b2 = c2; b3 = c3;
  }
  v8f accs[4] = {acc0, acc1, acc2, acc3};
  int cb = lane & 15, rb = (lane >> 4) * 8;
#pragma unroll
  for (int ct = 0; ct < 4; ++ct) {
    int col = col0 + ct * 16 + cb;
    int sec = col >> 9;           // 0=q 1=k 2=v1
    int inner = col & 511;
    int h = inner >> 6, d = inner & 63;
#pragma unroll
    for (int j = 0; j < 8; ++j) {
      int row = row0 + rb + j;
      int b = row / Nn, n = row % Nn;
      __bf16 val = (__bf16)accs[ct][j];
      long hoff = (long)b * HEADS + h;
      if (sec == 0)      qp[(hoff * NPAD + n) * DH + d] = val;
      else if (sec == 1) kp[(hoff * NPAD + n) * DH + d] = val;
      else               v1t[(hoff * DH + d) * MPAD + n] = val;
    }
  }
}

// ---------------------------------------------------------------------------
// 4) v2lin GEMM: [3152x512] @ [512x64] -> bf16 [BN][64]
// ---------------------------------------------------------------------------
__global__ void k_gemm_v2lin(const __bf16* __restrict__ A, const __bf16* __restrict__ Bt,
                             __bf16* __restrict__ v2lin) {
  int wave = (blockIdx.x * blockDim.x + threadIdx.x) >> 5;
  int lane = threadIdx.x & 31;
  const int MT = BN / 16;
  if (wave >= MT) return;
  int row0 = wave * 16;
  v8f acc[4] = {};
  const __bf16* arow = A + (long)row0 * DIMc;
#pragma unroll 4
  for (int k = 0; k < DIMc; k += 32) {
    v16bf a = load_a_frag(arow + k, DIMc);
#pragma unroll
    for (int ct = 0; ct < 4; ++ct)
      acc[ct] = wmma_bf16(a, load_b_frag(Bt + (long)(ct * 16) * DIMc + k, DIMc), acc[ct]);
  }
  int cb = lane & 15, rb = (lane >> 4) * 8;
#pragma unroll
  for (int ct = 0; ct < 4; ++ct) {
    int col = ct * 16 + cb;
#pragma unroll
    for (int j = 0; j < 8; ++j)
      v2lin[(long)(row0 + rb + j) * DRH + col] = (__bf16)acc[ct][j];
  }
}

// ---------------------------------------------------------------------------
// 5) RoPE on q/k (cls token passes through), bf16 in/out [b][h][NPAD][64]
// ---------------------------------------------------------------------------
__global__ void k_rope(const __bf16* __restrict__ qp, const __bf16* __restrict__ kp,
                       __bf16* __restrict__ qr, __bf16* __restrict__ kr) {
  int t = blockIdx.x * blockDim.x + threadIdx.x;
  if (t >= Bb * HEADS * Nn * DH) return;
  int d = t & 63;
  int n = (t >> 6) % Nn;
  int bh = t / (Nn * DH);
  long off = ((long)bh * NPAD + n) * DH + d;
  float qv = (float)qp[off], kv = (float)kp[off];
  if (n == 0) { qr[off] = (__bf16)qv; kr[off] = (__bf16)kv; return; }
  int p = n - 1;
  int x = p / Ww, y = p % Ww;
  int dd = d & 31;                      // emb[d] = freqs[d % 32]
  float pos = (dd < 16) ? (float)x : (float)y;
  int fj = dd & 15;
  float inv = __powf(10000.f, -((float)(2 * fj)) * (1.f / 32.f));
  float ang = pos * inv;
  float c = __cosf(ang), s = __sinf(ang);
  long off2 = (d < 32) ? off + 32 : off - 32;
  float sgn = (d < 32) ? -1.f : 1.f;
  float qo = (float)qp[off2], ko = (float)kp[off2];
  qr[off] = (__bf16)(qv * c + sgn * qo * s);
  kr[off] = (__bf16)(kv * c + sgn * ko * s);
}

// ---------------------------------------------------------------------------
// 6) Attention: block per (b,h); 8 waves, each owns 16-row strips.
//    v1t tile staged into LDS once per block (TDM if available, else async);
//    dots via WMMA; softmax in registers; attn strip staged in LDS;
//    out1 = attn @ v1 (A and B from LDS).
//    Dynamic LDS: [0,28672) = v1t tile, [28672, 28672+57344) = attn strips.
// ---------------------------------------------------------------------------
__global__ void k_attention(const __bf16* __restrict__ qr, const __bf16* __restrict__ kr,
                            const __bf16* __restrict__ v1t,
                            __bf16* __restrict__ attnw, __bf16* __restrict__ out1w) {
  extern __shared__ __align__(16) unsigned char dynsm[];
  __bf16* sv1t = (__bf16*)dynsm;                       // 64 x MPAD
  int bh = blockIdx.x;                    // 0..127
  int wave = threadIdx.x >> 5, lane = threadIdx.x & 31;
  __bf16* sattn = (__bf16*)(dynsm + 28672) + wave * (16 * MPAD);

  // stage the whole v1 tile for this (b,h): 28672 B
  {
    const char* gsrc = (const char*)(v1t + (long)bh * DH * MPAD);
#if HAVE_TDM
    if (threadIdx.x < 32) {               // one wave issues the TDM descriptor
      tdm_copy_1d(gsrc, dynsm, 28672);
      wait_tensor0();
    }
#else
    for (int c = threadIdx.x; c < 1792; c += 256)
      async_g2l_b128(gsrc + c * 16, dynsm + c * 16);
    wait_async0();
#endif
    __syncthreads();
  }

  const __bf16* q  = qr + (long)bh * NPAD * DH;
  const __bf16* kb = kr + (long)bh * NPAD * DH;
  int cb = lane & 15, rb = (lane >> 4) * 8;
  int b = bh >> 3, h = bh & 7;

  for (int s = wave; s < 13; s += 8) {
    int n0 = s * 16;
    v16bf a0 = load_a_frag(q + (long)n0 * DH + 0, DH);
    v16bf a1 = load_a_frag(q + (long)n0 * DH + 32, DH);
    v8f dots[13];
#pragma unroll
    for (int mt = 0; mt < 13; ++mt) {
      v8f acc = {};
      acc = wmma_bf16(a0, load_b_frag(kb + (long)(mt * 16) * DH + 0, DH), acc);
      acc = wmma_bf16(a1, load_b_frag(kb + (long)(mt * 16) * DH + 32, DH), acc);
      dots[mt] = acc;
    }
    // scale + mask out-of-range columns
#pragma unroll
    for (int mt = 0; mt < 13; ++mt) {
      bool oob = (mt * 16 + cb) >= Nn;
#pragma unroll
      for (int j = 0; j < 8; ++j) {
        float v = dots[mt][j] * 0.125f;
        dots[mt][j] = oob ? -1e30f : v;
      }
    }
    // softmax per row (rows live within a 16-lane half-group)
    float inv_s[8];
#pragma unroll
    for (int j = 0; j < 8; ++j) {
      float mx = dots[0][j];
#pragma unroll
      for (int mt = 1; mt < 13; ++mt) mx = fmaxf(mx, dots[mt][j]);
#pragma unroll
      for (int o = 1; o < 16; o <<= 1) mx = fmaxf(mx, __shfl_xor(mx, o));
      float sm = 0.f;
#pragma unroll
      for (int mt = 0; mt < 13; ++mt) {
        float e = __expf(dots[mt][j] - mx);
        dots[mt][j] = e;
        sm += e;
      }
#pragma unroll
      for (int o = 1; o < 16; o <<= 1) sm += __shfl_xor(sm, o);
      inv_s[j] = 1.f / sm;
    }
    // write attn strip (LDS, zero-padded to MPAD) + global copy for v2 kernel
#pragma unroll
    for (int mt = 0; mt < 14; ++mt) {
#pragma unroll
      for (int j = 0; j < 8; ++j) {
        int col = mt * 16 + cb;
        float v = (mt < 13 && col < Nn) ? dots[mt][j] * inv_s[j] : 0.f;
        int r = rb + j;
        sattn[r * MPAD + col] = (__bf16)v;
        int nrow = n0 + r;
        if (nrow < Nn && col < NPAD)
          attnw[((long)bh * Nn + nrow) * NPAD + col] = (__bf16)v;
      }
    }
    // out1 = attn @ v1   (both operands from LDS)
#pragma unroll
    for (int ct = 0; ct < 4; ++ct) {
      v8f acc = {};
#pragma unroll
      for (int kk = 0; kk < MPAD; kk += 32) {
        v16bf a = load_a_frag(sattn + kk, MPAD);
        v16bf bf = load_b_frag(sv1t + (long)(ct * 16) * MPAD + kk, MPAD);
        acc = wmma_bf16(a, bf, acc);
      }
      int col = ct * 16 + cb;
#pragma unroll
      for (int j = 0; j < 8; ++j) {
        int nrow = n0 + rb + j;
        if (nrow < Nn)
          out1w[((long)b * Nn + nrow) * INNER + h * DH + col] = (__bf16)acc[j];
      }
    }
  }
}

// ---------------------------------------------------------------------------
// 7) Translution path: block per n, 4 waves cooperate per m (wave = col-tile).
//    relT[IDX(n,m)] double-buffered in LDS via TDM (or async fallback);
//    attn slice async-staged row-contiguous. No cross-wave reduction needed.
//    Dynamic LDS: [0,53248) attn rows [bh][NPAD], [53248, +16384) rel buffers.
// ---------------------------------------------------------------------------
__global__ void k_v2(const __bf16* __restrict__ v2lin, const __bf16* __restrict__ relT,
                     const __bf16* __restrict__ attnw, __bf16* __restrict__ out2w) {
  extern __shared__ __align__(16) unsigned char dynsm[];
  __bf16* sa2  = (__bf16*)dynsm;                        // [128][NPAD]
  __bf16* srel = (__bf16*)(dynsm + 53248);              // 2 x [64][64]
  int n = blockIdx.x;                                   // 0..196
  int w = threadIdx.x >> 5, lane = threadIdx.x & 31;    // 4 waves, ct = w
  int cb = lane & 15, rb = (lane >> 4) * 8;

  // stage attn[:, :, n, :] rows: 128 rows x 416 B = 3328 x b128 (async path)
  for (int c = threadIdx.x; c < 3328; c += 128) {
    int row = c / 26, cc = c - row * 26;
    const char* src = (const char*)(attnw + ((long)row * Nn + n) * NPAD) + cc * 16;
    async_g2l_b128(src, (char*)sa2 + row * 416 + cc * 16);
  }
  // stage rel tile for m=0 into buffer 0: 8192 B
  {
    const char* src = (const char*)(relT + (long)rel_index(n, 0) * (DRH * DRH));
#if HAVE_TDM
    if (w == 0) tdm_copy_1d(src, srel, 8192);
#else
    for (int c = threadIdx.x; c < 512; c += 128)
      async_g2l_b128(src + c * 16, (char*)srel + c * 16);
#endif
  }
  wait_async0();
#if HAVE_TDM
  if (w == 0) wait_tensor0();
#endif
  __syncthreads();

  v8f sums = {};
  for (int m = 0; m < Nn; ++m) {
    __bf16* cur = srel + (m & 1) * (DRH * DRH);
    if (m + 1 < Nn) {
      const char* src = (const char*)(relT + (long)rel_index(n, m + 1) * (DRH * DRH));
      char* dst = (char*)(srel + ((m + 1) & 1) * (DRH * DRH));
#if HAVE_TDM
      if (w == 0) tdm_copy_1d(src, dst, 8192);
#else
      for (int c = threadIdx.x; c < 512; c += 128)
        async_g2l_b128(src + c * 16, dst + c * 16);
#endif
    }
    // t = v2lin[:, m, :] @ rel (this wave's 16-col tile)
    const __bf16* Ab = v2lin + (long)m * DRH;           // rows are b, lda = Nn*DRH
    v16bf a0 = load_a_frag(Ab + 0, Nn * DRH);
    v16bf a1 = load_a_frag(Ab + 32, Nn * DRH);
    v8f t = {};
    t = wmma_bf16(a0, load_b_frag(cur + (long)(w * 16) * DRH + 0, DRH), t);
    t = wmma_bf16(a1, load_b_frag(cur + (long)(w * 16) * DRH + 32, DRH), t);
    int col = w * 16 + cb;
    int h = col >> 3;
#pragma unroll
    for (int j = 0; j < 8; ++j) {
      float av = (float)sa2[(rb + j) * 8 * NPAD + h * NPAD + m];
      sums[j] += t[j] * av;
    }
#if HAVE_TDM
    if (w == 0) wait_tensor0();
#else
    wait_async0();
#endif
    __syncthreads();
  }
  int col = w * 16 + cb;
#pragma unroll
  for (int j = 0; j < 8; ++j)
    out2w[((long)(rb + j) * Nn + n) * DRH + col] = (__bf16)sums[j];
}

// ---------------------------------------------------------------------------
// 8) F = 0.5 * (out1 + out2 @ w_v3)   -> bf16 [BN][512]
// ---------------------------------------------------------------------------
__global__ void k_f1(const __bf16* __restrict__ out2w, const __bf16* __restrict__ wv3T,
                     const __bf16* __restrict__ out1w, __bf16* __restrict__ Fw) {
  int wave = (blockIdx.x * blockDim.x + threadIdx.x) >> 5;
  int lane = threadIdx.x & 31;
  const int MT = BN / 16, NT = INNER / 64;           // 197 x 8
  if (wave >= MT * NT) return;
  int mt = wave / NT, nt = wave % NT;
  int row0 = mt * 16, col0 = nt * 64;
  v8f acc[4] = {};
#pragma unroll
  for (int k = 0; k < DRH; k += 32) {
    v16bf a = load_a_frag(out2w + (long)row0 * DRH + k, DRH);
#pragma unroll
    for (int ct = 0; ct < 4; ++ct)
      acc[ct] = wmma_bf16(a, load_b_frag(wv3T + (long)(col0 + ct * 16) * DRH + k, DRH), acc[ct]);
  }
  int cb = lane & 15, rb = (lane >> 4) * 8;
#pragma unroll
  for (int ct = 0; ct < 4; ++ct) {
    int col = col0 + ct * 16 + cb;
#pragma unroll
    for (int j = 0; j < 8; ++j) {
      long row = row0 + rb + j;
      float v = 0.5f * (acc[ct][j] + (float)out1w[row * INNER + col]);
      Fw[row * INNER + col] = (__bf16)v;
    }
  }
}

// ---------------------------------------------------------------------------
// 9) out = F @ w_out + b_out   -> fp32 [BN][512]  (software-pipelined)
// ---------------------------------------------------------------------------
__global__ void k_f2(const __bf16* __restrict__ Fw, const __bf16* __restrict__ woutT,
                     const float* __restrict__ bout, float* __restrict__ out) {
  int wave = (blockIdx.x * blockDim.x + threadIdx.x) >> 5;
  int lane = threadIdx.x & 31;
  const int MT = BN / 16, NT = DIMc / 64;            // 197 x 8
  if (wave >= MT * NT) return;
  int mt = wave / NT, nt = wave % NT;
  int row0 = mt * 16, col0 = nt * 64;
  v8f acc0 = {}, acc1 = {}, acc2 = {}, acc3 = {};
  const __bf16* arow = Fw + (long)row0 * INNER;
  const __bf16* bp = woutT + (long)col0 * INNER;
  v16bf a  = load_a_frag(arow, INNER);
  v16bf b0 = load_b_frag(bp + 0L * 16 * INNER, INNER);
  v16bf b1 = load_b_frag(bp + 1L * 16 * INNER, INNER);
  v16bf b2 = load_b_frag(bp + 2L * 16 * INNER, INNER);
  v16bf b3 = load_b_frag(bp + 3L * 16 * INNER, INNER);
#pragma unroll
  for (int k = 0; k < INNER; k += 32) {
    v16bf an = a, c0 = b0, c1 = b1, c2 = b2, c3 = b3;
    if (k + 32 < INNER) {
      __builtin_prefetch(arow + k + 96, 0, 0);
      an = load_a_frag(arow + k + 32, INNER);
      c0 = load_b_frag(bp + 0L * 16 * INNER + k + 32, INNER);
      c1 = load_b_frag(bp + 1L * 16 * INNER + k + 32, INNER);
      c2 = load_b_frag(bp + 2L * 16 * INNER + k + 32, INNER);
      c3 = load_b_frag(bp + 3L * 16 * INNER + k + 32, INNER);
    }
    acc0 = wmma_bf16(a, b0, acc0);
    acc1 = wmma_bf16(a, b1, acc1);
    acc2 = wmma_bf16(a, b2, acc2);
    acc3 = wmma_bf16(a, b3, acc3);
    a = an; b0 = c0; b1 = c1; b2 = c2; b3 = c3;
  }
  int cb = lane & 15, rb = (lane >> 4) * 8;
  v8f accs[4] = {acc0, acc1, acc2, acc3};
#pragma unroll
  for (int ct = 0; ct < 4; ++ct) {
    int col = col0 + ct * 16 + cb;
    float bv = bout[col];
#pragma unroll
    for (int j = 0; j < 8; ++j) {
      long row = row0 + rb + j;
      out[row * DIMc + col] = accs[ct][j] + bv;
    }
  }
}

// ---------------------------------------------------------------------------
// host launch
// ---------------------------------------------------------------------------
extern "C" void kernel_launch(void* const* d_in, const int* in_sizes, int n_in,
                              void* d_out, int out_size, void* d_ws, size_t ws_size,
                              hipStream_t stream) {
  const float* x    = (const float*)d_in[0];
  const float* lng  = (const float*)d_in[1];
  const float* lnb  = (const float*)d_in[2];
  const float* wqkv = (const float*)d_in[3];
  const float* wv1  = (const float*)d_in[4];
  const float* rel  = (const float*)d_in[5];
  const float* wv3  = (const float*)d_in[6];
  const float* wout = (const float*)d_in[7];
  const float* bout = (const float*)d_in[8];
  float* out = (float*)d_out;

  char* ws = (char*)d_ws;
  size_t off = 0;
  auto alloc = [&](size_t bytes) -> char* {
    char* p = ws + off;
    off = (off + bytes + 255) & ~(size_t)255;
    return p;
  };
  __bf16* XN    = (__bf16*)alloc((size_t)BN * DIMc * 2);
  __bf16* WQKVT = (__bf16*)alloc((size_t)3 * INNER * DIMc * 2);
  __bf16* WV1T  = (__bf16*)alloc((size_t)DRH * DIMc * 2);
  __bf16* RELT  = (__bf16*)alloc((size_t)NUM_REL * DRH * DRH * 2);
  __bf16* WV3T  = (__bf16*)alloc((size_t)INNER * DRH * 2);
  __bf16* WOUTT = (__bf16*)alloc((size_t)DIMc * INNER * 2);
  __bf16* QP    = (__bf16*)alloc((size_t)Bb * HEADS * NPAD * DH * 2);
  __bf16* KP    = (__bf16*)alloc((size_t)Bb * HEADS * NPAD * DH * 2);
  __bf16* QR    = (__bf16*)alloc((size_t)Bb * HEADS * NPAD * DH * 2);
  __bf16* KR    = (__bf16*)alloc((size_t)Bb * HEADS * NPAD * DH * 2);
  __bf16* V1T   = (__bf16*)alloc((size_t)Bb * HEADS * DH * MPAD * 2);
  __bf16* V2LIN = (__bf16*)alloc((size_t)BN * DRH * 2);
  __bf16* ATTNW = (__bf16*)alloc((size_t)Bb * HEADS * Nn * NPAD * 2);
  __bf16* OUT1W = (__bf16*)alloc((size_t)BN * INNER * 2);
  __bf16* OUT2W = (__bf16*)alloc((size_t)BN * DRH * 2);
  __bf16* FW    = (__bf16*)alloc((size_t)BN * INNER * 2);

  // zero workspace (provides zero padding for all padded tiles)
  hipMemsetAsync(d_ws, 0, off, stream);

  auto tgrid = [](long total) {
    long b = (total + 255) / 256;
    if (b > 65535) b = 65535;
    return (int)b;
  };
  // weight converts/transposes
  k_transpose<<<tgrid(512L * 1536), 256, 0, stream>>>(wqkv, WQKVT, 1, DIMc, 3 * INNER);
  k_transpose<<<tgrid(512L * 64),   256, 0, stream>>>(wv1,  WV1T,  1, DIMc, DRH);
  k_transpose<<<tgrid((long)NUM_REL * 64 * 64), 256, 0, stream>>>(rel, RELT, NUM_REL, DRH, DRH);
  k_transpose<<<tgrid(64L * 512),   256, 0, stream>>>(wv3,  WV3T,  1, DRH, INNER);
  k_transpose<<<tgrid(512L * 512),  256, 0, stream>>>(wout, WOUTT, 1, INNER, DIMc);

  // layernorm
  k_layernorm<<<BN, 256, 0, stream>>>(x, lng, lnb, XN);

  // qkv + v2lin projections
  {
    int waves = (BN / 16) * ((3 * INNER) / 64);
    k_gemm_qkv<<<(waves * 32 + 255) / 256, 256, 0, stream>>>(XN, WQKVT, QP, KP, V1T);
  }
  {
    int waves = BN / 16;
    k_gemm_v2lin<<<(waves * 32 + 255) / 256, 256, 0, stream>>>(XN, WV1T, V2LIN);
  }

  // RoPE
  {
    long tot = (long)Bb * HEADS * Nn * DH;
    k_rope<<<(int)((tot + 255) / 256), 256, 0, stream>>>(QP, KP, QR, KR);
  }

  // attention (dots + softmax + out1), dynamic LDS = 28672 + 57344
  k_attention<<<Bb * HEADS, 256, 86016, stream>>>(QR, KR, V1T, ATTNW, OUT1W);

  // translution path, dynamic LDS = 53248 + 16384
  k_v2<<<Nn, 128, 69632, stream>>>(V2LIN, RELT, ATTNW, OUT2W);

  // combine + output projection
  {
    int waves = (BN / 16) * (INNER / 64);
    k_f1<<<(waves * 32 + 255) / 256, 256, 0, stream>>>(OUT2W, WV3T, OUT1W, FW);
    k_f2<<<(waves * 32 + 255) / 256, 256, 0, stream>>>(FW, WOUTT, bout, out);
  }
}
